// MultiheadAttention_19550691132057
// MI455X (gfx1250) — matrix operand order repaired and checked by reading, not compiled, not run on hardware
//
#include <hip/hip_runtime.h>
#include <hip/hip_bf16.h>

typedef __attribute__((ext_vector_type(16))) _Float16 v16h;
typedef __attribute__((ext_vector_type(8)))  _Float16 v8h;
typedef __attribute__((ext_vector_type(8)))  float    v8f;
typedef __attribute__((ext_vector_type(4)))  unsigned v4u;
typedef __attribute__((ext_vector_type(8)))  unsigned v8u;

#define SLEN 2048
#define EMB  1024
#define DHD  64
#define NHEAD 16

__device__ __forceinline__ v8f wmma_f16(v16h a, v16h b, v8f c) {
    // D = A(16x32 f16) * B(32x16 f16) + C(16x16 f32)
    return __builtin_amdgcn_wmma_f32_16x16x32_f16(
        /*neg_a=*/false, a, /*neg_b=*/false, b,
        /*c_mod=*/(short)0, c, /*reuse_a=*/false, /*reuse_b=*/false);
}

// A-fragment: elements 0..7 from p0 (K = base..base+7), 8..15 from p1 (K = base+8..base+15)
__device__ __forceinline__ v16h ld_a_frag(const _Float16* p0, const _Float16* p1) {
    v8h lo = *(const v8h*)p0;
    v8h hi = *(const v8h*)p1;
    v16h r;
#pragma unroll
    for (int i = 0; i < 8; ++i) { r[i] = lo[i]; r[i + 8] = hi[i]; }
    return r;
}

__device__ __forceinline__ v16h ld16(const _Float16* p) {
    return *(const v16h*)p;
}

// CDNA5 async copy: 16 bytes global -> LDS, tracked by ASYNCcnt (no VGPR data path)
__device__ __forceinline__ void async_copy_b128(unsigned lds_off, const void* gptr) {
    asm volatile("global_load_async_to_lds_b128 %0, %1, off"
                 :: "v"(lds_off), "v"(gptr) : "memory");
}
__device__ __forceinline__ void wait_async0() {
    asm volatile("s_wait_asynccnt 0x0" ::: "memory");
}

// CDNA5 Tensor Data Mover: one instruction moves a 2-D tile global -> LDS.
// g0 = D# group0 (4 SGPRs), g1 = D# group1 (8 SGPRs); VADDR2/3 omitted (<=2D).
__device__ __forceinline__ void tdm_load_2d(v4u g0, v8u g1) {
    asm volatile("tensor_load_to_lds %0, %1" :: "s"(g0), "s"(g1) : "memory");
}
__device__ __forceinline__ void wait_tensor0() {
    __builtin_amdgcn_s_wait_tensorcnt((short)0);
}

// ---------------------------------------------------------------------------
// Kernel 1: per-head input projection.  GEMM M=4096, N=1024 (h*64+d), K=1024.
// Wave tile 32x32 (4 wmma / k-step), WG tile 128x64 (8 waves, 4x2).
// ---------------------------------------------------------------------------
__global__ __launch_bounds__(256) void proj_kernel(
    const float* __restrict__ X,    // [4096,1024] f32
    const float* __restrict__ W,    // [H,E,DH] f32
    const float* __restrict__ bias, // [1024] f32 (H*DH flat)
    _Float16* __restrict__ Out)     // [B*H, S, DH] f16
{
    const int tid  = threadIdx.x;
    const int wave = tid >> 5;
    const int lane = tid & 31;
    const int ln   = lane & 15;
    const int lh   = lane >> 4;
    const int m0   = blockIdx.x * 128;
    const int n0   = blockIdx.y * 64;
    const int waveR = wave >> 1, waveC = wave & 1;   // 4 x 2 wave grid

    __shared__ __align__(32) _Float16 Xs[128][32];
    __shared__ __align__(32) _Float16 Ws[64][32];    // transposed: Ws[n][k]

    v8f acc[2][2] = {};
    for (int kk = 0; kk < EMB; kk += 32) {
        __syncthreads();
        { // stage X tile (128x32), 16 halfs per thread
            int r = tid >> 1, c = (tid & 1) * 16;
            const float* src = X + (size_t)(m0 + r) * EMB + kk + c;
#pragma unroll
            for (int i = 0; i < 16; ++i) Xs[r][c + i] = (_Float16)src[i];
        }
        { // stage W tile transposed (64x32), 8 per thread
            int n = tid >> 2, ks = (tid & 3) * 8;
            int ng = n0 + n, h = ng >> 6, d = ng & 63;
            const float* src = W + (size_t)h * (EMB * DHD) + (size_t)(kk + ks) * DHD + d;
#pragma unroll
            for (int i = 0; i < 8; ++i) Ws[n][ks + i] = (_Float16)src[(size_t)i * DHD];
        }
        __syncthreads();
        const int ar0 = waveR * 32 + ln;
        v16h a0 = ld_a_frag(&Xs[ar0][lh * 16],      &Xs[ar0][lh * 16 + 8]);
        v16h a1 = ld_a_frag(&Xs[ar0 + 16][lh * 16], &Xs[ar0 + 16][lh * 16 + 8]);
        v16h b0 = ld16(&Ws[waveC * 32 + ln][lh * 16]);
        v16h b1 = ld16(&Ws[waveC * 32 + 16 + ln][lh * 16]);
        acc[0][0] = wmma_f16(a0, b0, acc[0][0]);
        acc[0][1] = wmma_f16(a0, b1, acc[0][1]);
        acc[1][0] = wmma_f16(a1, b0, acc[1][0]);
        acc[1][1] = wmma_f16(a1, b1, acc[1][1]);
    }

#pragma unroll
    for (int ct = 0; ct < 2; ++ct) {
        int ng = n0 + waveC * 32 + ct * 16 + ln;
        float bvv = bias[ng];
        int h = ng >> 6, d = ng & 63;
#pragma unroll
        for (int rt = 0; rt < 2; ++rt) {
#pragma unroll
            for (int r = 0; r < 8; ++r) {
                int m = m0 + waveR * 32 + rt * 16 + r + 8 * lh;
                int b = m >> 11, s = m & (SLEN - 1);
                Out[(((size_t)(b * NHEAD + h)) * SLEN + s) * DHD + d] =
                    (_Float16)(acc[rt][ct][r] + bvv);
            }
        }
    }
}

// ---------------------------------------------------------------------------
// Kernel 2: flash attention per (b,h).  WG = 128 query rows (8 waves x 16).
// Q block staged via async global->LDS copies; K tile via TDM tensor_load_to_lds;
// V transpose-staged manually.
// ---------------------------------------------------------------------------
__global__ __launch_bounds__(256) void attn_kernel(
    const _Float16* __restrict__ Q,
    const _Float16* __restrict__ K,
    const _Float16* __restrict__ V,
    _Float16* __restrict__ O)       // [B, S, E] f16 (head-concat)
{
    const int blk  = blockIdx.x;      // B*H*(S/128) = 512
    const int qblk = blk & 15;        // S/128 = 16
    const int bh   = blk >> 4;        // 0..31
    const int tid  = threadIdx.x;
    const int wave = tid >> 5;
    const int lane = tid & 31;
    const int ln   = lane & 15;
    const int lh   = lane >> 4;

    const size_t bhoff = (size_t)bh * SLEN * DHD;
    const _Float16* Qb = Q + bhoff;
    const _Float16* Kb = K + bhoff;
    const _Float16* Vb = V + bhoff;

    __shared__ __align__(32) _Float16 Qblk[128][DHD];  // this WG's query rows
    __shared__ __align__(32) _Float16 Ks[32][DHD];     // K tile row-major (== B^T)
    __shared__ __align__(32) _Float16 Vt[DHD][32];     // V tile transposed (d-major)
    __shared__ __align__(32) _Float16 Pl[8][16][32];   // per-wave P (row-major)

    const unsigned qBase  = (unsigned)(unsigned long long)&Qblk[0][0];
    const unsigned ksBase = (unsigned)(unsigned long long)&Ks[0][0];

    // --- one-time Q staging: 128x64 f16 = 16KB via async copies (4 x b128/thread)
    {
        const _Float16* qsrc = Qb + (size_t)qblk * 128 * DHD;
#pragma unroll
        for (int i = 0; i < 4; ++i) {
            size_t e = ((size_t)tid + (size_t)i * 256) * 8;
            async_copy_b128(qBase + (unsigned)(e * 2), (const void*)(qsrc + e));
        }
    }
    wait_async0();
    __syncthreads();

    const int kb = lh * 16;
    const _Float16* qp = &Qblk[wave * 16 + ln][0];     // this lane's A-frag row
    v16h qa0 = ld_a_frag(qp + 0  + kb, qp + 8 + kb);   // head_dim 0..31
    v16h qa1 = ld_a_frag(qp + 32 + kb, qp + 40 + kb);  // head_dim 32..63

    // --- loop-invariant TDM descriptor group1: 2-byte elems, tensor 64 x 2048
    //     (stride 64), tile 64 x 32  => one K tile per instruction
    v8u tg1;
    tg1[0] = 1u << 16;        // workgroup_mask=0, data_size=1 (2 bytes)
    tg1[1] = 64u << 16;       // tensor_dim0 = 64   (bits 79:48, low16)
    tg1[2] = 2048u << 16;     // dim0 hi = 0; tensor_dim1 = 2048 (bits 111:80, low16)
    tg1[3] = 64u << 16;       // dim1 hi = 0; tile_dim0 = 64 (bits 127:112)
    tg1[4] = 32u;             // tile_dim1 = 32; tile_dim2 = 0
    tg1[5] = 64u;             // tensor_dim0_stride = 64
    tg1[6] = 0u;              // stride0 hi, stride1 lo
    tg1[7] = 0u;              // stride1 hi

    v8f acc0 = {}, acc1 = {}, acc2 = {}, acc3 = {};
    float mrow[8], lrow[8];
#pragma unroll
    for (int r = 0; r < 8; ++r) { mrow[r] = -1e30f; lrow[r] = 0.f; }

    const float SCALE = 0.03125f;  // 1/sqrt(1024)

    for (int k0 = 0; k0 < SLEN; k0 += 32) {
        __syncthreads();
        // --- K tile (32x64 f16 = 4KB) via Tensor Data Mover, issued by wave 0
        if (wave == 0) {
            unsigned long long ga = (unsigned long long)(Kb + (size_t)k0 * DHD);
            v4u g0;
            g0[0] = 1u;                                   // count=1 (valid user D#)
            g0[1] = ksBase;                               // lds_addr
            g0[2] = (unsigned)ga;                         // global_addr[31:0]
            g0[3] = (unsigned)((ga >> 32) & 0x01FFFFFFu)  // global_addr[56:32]
                  | (2u << 30);                           // type = 2 ("image")
            tdm_load_2d(g0, tg1);
        }
        { // manual transpose-stage V tile: 8 contiguous halfs -> scattered ds stores
            int idx = tid * 8;
            int r = idx >> 6, d = idx & 63;
            const _Float16* src = Vb + (size_t)(k0 + r) * DHD + d;
            v8h vv = *(const v8h*)src;
#pragma unroll
            for (int i = 0; i < 8; ++i) Vt[d + i][r] = vv[i];
        }
        { // prefetch next tiles into cache while this block computes
            size_t nk = (size_t)((k0 + 32) & (SLEN - 1)) * DHD + (size_t)tid * 8;
            __builtin_prefetch(Kb + nk, 0, 3);
            __builtin_prefetch(Vb + nk, 0, 3);
        }
        if (wave == 0) wait_tensor0();
        __syncthreads();

        // scores: two 16-key tiles; Ks rows are B^T (contiguous 32B LDS loads)
        v8f s0 = {}, s1 = {};
        {
            const int koff = lh * 16;
            v16h b00 = ld16(&Ks[ln][koff]);
            v16h b01 = ld16(&Ks[ln][32 + koff]);
            v16h b10 = ld16(&Ks[16 + ln][koff]);
            v16h b11 = ld16(&Ks[16 + ln][32 + koff]);
            s0 = wmma_f16(qa0, b00, s0);
            s0 = wmma_f16(qa1, b01, s0);
            s1 = wmma_f16(qa0, b10, s1);
            s1 = wmma_f16(qa1, b11, s1);
        }

        // online softmax; lane owns rows r+8*lh in C layout, one elem per key tile
#pragma unroll
        for (int r = 0; r < 8; ++r) {
            float v0 = s0[r] * SCALE;
            float v1 = s1[r] * SCALE;
            float mx = fmaxf(v0, v1);
#pragma unroll
            for (int off = 1; off < 16; off <<= 1)
                mx = fmaxf(mx, __shfl_xor(mx, off, 32));
            float mn = fmaxf(mrow[r], mx);
            float alpha = __expf(mrow[r] - mn);
            float p0 = __expf(v0 - mn);
            float p1 = __expf(v1 - mn);
            float ssum = p0 + p1;
#pragma unroll
            for (int off = 1; off < 16; off <<= 1)
                ssum += __shfl_xor(ssum, off, 32);
            lrow[r] = lrow[r] * alpha + ssum;
            mrow[r] = mn;
            acc0[r] *= alpha; acc1[r] *= alpha;
            acc2[r] *= alpha; acc3[r] *= alpha;
            int row = r + 8 * lh;
            Pl[wave][row][ln]      = (_Float16)p0;
            Pl[wave][row][16 + ln] = (_Float16)p1;
        }

        // P(16x32) x V(32x64): A-frag from per-wave LDS, B-frags from Vt
        v16h pa = ld_a_frag(&Pl[wave][ln][kb], &Pl[wave][ln][8 + kb]);
        {
            const int koff2 = lh * 16;
            v16h vb0 = ld16(&Vt[ 0 + ln][koff2]);
            v16h vb1 = ld16(&Vt[16 + ln][koff2]);
            v16h vb2 = ld16(&Vt[32 + ln][koff2]);
            v16h vb3 = ld16(&Vt[48 + ln][koff2]);
            acc0 = wmma_f16(pa, vb0, acc0);
            acc1 = wmma_f16(pa, vb1, acc1);
            acc2 = wmma_f16(pa, vb2, acc2);
            acc3 = wmma_f16(pa, vb3, acc3);
        }
    }

    // normalize and store head-concat output: O[b][s][h*64 + d]
    const int b = bh >> 4, h = bh & 15;
    _Float16* Ob = O + ((size_t)b * SLEN) * EMB + (size_t)h * DHD;
    const int srow = qblk * 128 + wave * 16;
#pragma unroll
    for (int r = 0; r < 8; ++r) {
        int row = srow + r + 8 * lh;
        float inv = 1.0f / lrow[r];
        Ob[(size_t)row * EMB +  0 + ln] = (_Float16)(acc0[r] * inv);
        Ob[(size_t)row * EMB + 16 + ln] = (_Float16)(acc1[r] * inv);
        Ob[(size_t)row * EMB + 32 + ln] = (_Float16)(acc2[r] * inv);
        Ob[(size_t)row * EMB + 48 + ln] = (_Float16)(acc3[r] * inv);
    }
}

// ---------------------------------------------------------------------------
// Kernel 3: output projection  out = A(4096x1024 f16) x Wo(1024x1024 f32) + bo
// Wave tile 32x32 (4 wmma / k-step), WG tile 128x64.  A direct from global.
// ---------------------------------------------------------------------------
__global__ __launch_bounds__(256) void oproj_kernel(
    const _Float16* __restrict__ A,
    const float* __restrict__ Wo,
    const float* __restrict__ bo,
    float* __restrict__ Out)
{
    const int tid  = threadIdx.x;
    const int wave = tid >> 5;
    const int lane = tid & 31;
    const int ln   = lane & 15;
    const int lh   = lane >> 4;
    const int m0   = blockIdx.x * 128;
    const int n0   = blockIdx.y * 64;
    const int waveR = wave >> 1, waveC = wave & 1;

    __shared__ __align__(32) _Float16 Ws[64][32];  // transposed: Ws[n][k]

    v8f acc[2][2] = {};
    const int ar0 = m0 + waveR * 32 + ln;
    const _Float16* ap0 = A + (size_t)ar0 * EMB;
    const _Float16* ap1 = A + (size_t)(ar0 + 16) * EMB;

    for (int kk = 0; kk < EMB; kk += 32) {
        __syncthreads();
        { // stage Wo tile transposed (64x32), 8 per thread
            int n = tid >> 2, ks = (tid & 3) * 8;
            const float* src = Wo + (size_t)(kk + ks) * EMB + n0 + n;
#pragma unroll
            for (int i = 0; i < 8; ++i) Ws[n][ks + i] = (_Float16)src[(size_t)i * EMB];
        }
        __syncthreads();
        v16h a0 = ld_a_frag(ap0 + kk + lh * 16, ap0 + kk + lh * 16 + 8);
        v16h a1 = ld_a_frag(ap1 + kk + lh * 16, ap1 + kk + lh * 16 + 8);
        v16h b0 = ld16(&Ws[waveC * 32 + ln][lh * 16]);
        v16h b1 = ld16(&Ws[waveC * 32 + 16 + ln][lh * 16]);
        acc[0][0] = wmma_f16(a0, b0, acc[0][0]);
        acc[0][1] = wmma_f16(a0, b1, acc[0][1]);
        acc[1][0] = wmma_f16(a1, b0, acc[1][0]);
        acc[1][1] = wmma_f16(a1, b1, acc[1][1]);
    }

#pragma unroll
    for (int ct = 0; ct < 2; ++ct) {
        int ng = n0 + waveC * 32 + ct * 16 + ln;
        float bvv = bo[ng];
#pragma unroll
        for (int rt = 0; rt < 2; ++rt) {
#pragma unroll
            for (int r = 0; r < 8; ++r) {
                int m = m0 + waveR * 32 + rt * 16 + r + 8 * lh;
                Out[(size_t)m * EMB + ng] = acc[rt][ct][r] + bvv;
            }
        }
    }
}

// ---------------------------------------------------------------------------
extern "C" void kernel_launch(void* const* d_in, const int* in_sizes, int n_in,
                              void* d_out, int out_size, void* d_ws, size_t ws_size,
                              hipStream_t stream) {
    (void)in_sizes; (void)n_in; (void)out_size; (void)ws_size;
    const float* query = (const float*)d_in[0];
    const float* key   = (const float*)d_in[1];
    const float* value = (const float*)d_in[2];
    const float* Wq    = (const float*)d_in[3];
    const float* bq    = (const float*)d_in[4];
    const float* Wk    = (const float*)d_in[5];
    const float* bk    = (const float*)d_in[6];
    const float* Wv    = (const float*)d_in[7];
    const float* bv    = (const float*)d_in[8];
    const float* Wo    = (const float*)d_in[9];
    const float* bo    = (const float*)d_in[10];
    float* out = (float*)d_out;

    const size_t NELEM = (size_t)4096 * 1024;
    _Float16* Qh = (_Float16*)d_ws;
    _Float16* Kh = Qh + NELEM;
    _Float16* Vh = Kh + NELEM;
    _Float16* Oh = Vh + NELEM;

    dim3 pgrid(32, 16);   // 4096/128 x 1024/64
    proj_kernel<<<pgrid, 256, 0, stream>>>(query, Wq, bq, Qh);
    proj_kernel<<<pgrid, 256, 0, stream>>>(key,   Wk, bk, Kh);
    proj_kernel<<<pgrid, 256, 0, stream>>>(value, Wv, bv, Vh);

    attn_kernel<<<dim3(512), 256, 0, stream>>>(Qh, Kh, Vh, Oh);

    oproj_kernel<<<pgrid, 256, 0, stream>>>(Oh, Wo, bo, out);
}
